// SelfAttention_36223754174848
// MI455X (gfx1250) — matrix operand order repaired and checked by reading, not compiled, not run on hardware
//
#include <hip/hip_runtime.h>
#include <hip/hip_bf16.h>

typedef __attribute__((ext_vector_type(16))) __bf16 v16bf;
typedef __attribute__((ext_vector_type(8)))  float  v8f;
typedef int v4i_ __attribute__((vector_size(16)));   // matches async builtin param

#define DIM   128
#define LAYER 32
#define NTOK  4096
#define BATCH 8
#define FPAD  40    // attn LDS pitch (80B rows)
#define WPAD  136   // proj LDS pitch (272B rows, 16B aligned, conflict-free)

#if defined(__has_builtin)
# if __has_builtin(__builtin_amdgcn_global_load_async_to_lds_b128)
#  define HAVE_ASYNC_LDS 1
# endif
#endif

#ifdef HAVE_ASYNC_LDS
__device__ __forceinline__ void async_cp_b128(const void* g, void* l) {
    __builtin_amdgcn_global_load_async_to_lds_b128(
        (__attribute__((address_space(1))) v4i_*)(unsigned long long)g,
        (__attribute__((address_space(3))) v4i_*)(unsigned)(unsigned long long)l,
        0, 0);
}
__device__ __forceinline__ void async_wait0() {
# if __has_builtin(__builtin_amdgcn_s_wait_asynccnt)
    __builtin_amdgcn_s_wait_asynccnt(0);
# else
    asm volatile("s_wait_asynccnt 0x0" ::: "memory");
# endif
}
#endif

__device__ __forceinline__ float fast_exp2(float x) {
#if defined(__has_builtin)
# if __has_builtin(__builtin_amdgcn_exp2f)
    return __builtin_amdgcn_exp2f(x);
# else
    return exp2f(x);
# endif
#else
    return exp2f(x);
#endif
}

// cross-half (lane i <-> lane i+16) exchange via v_permlanex16, identity map
__device__ __forceinline__ unsigned permx16_u32(unsigned v) {
    return (unsigned)__builtin_amdgcn_permlanex16(
        (int)v, (int)v, 0x76543210, (int)0xFEDCBA98, false, false);
}
__device__ __forceinline__ float permx16_f32(float v) {
    return __int_as_float((int)permx16_u32((unsigned)__float_as_int(v)));
}

union frag16 { uint4 u[2]; unsigned w32[8]; v16bf v; };
union pk2    { struct { __bf16 lo, hi; } s; unsigned u; };
union pk8    { uint4 q; unsigned w[4]; };

#define LOG2E 1.44269504088896340736f

// ---------------------------------------------------------------------------
// Phase 1: 1x1 conv projections via WMMA.
//  out[o][n] = sum_c W[o][c] * X[c][n]  with A=W (16x32), B=X^T rows (32x16).
//  Emits f,g:[B,N,32] bf16 (G pre-scaled by log2e) and hT:[B,128,N] bf16.
// ---------------------------------------------------------------------------
__global__ __launch_bounds__(128) void proj_kernel(
    const float* __restrict__ x,
    const float* __restrict__ WF, const float* __restrict__ WG,
    const float* __restrict__ WH,
    __bf16* __restrict__ f_ws, __bf16* __restrict__ g_ws,
    __bf16* __restrict__ h_wsT)
{
    __shared__ __align__(16) __bf16 xsT[128][WPAD];  // [n][c]  ~34 KB
    __shared__ __align__(16) __bf16 Wb[192][WPAD];   // rows: 0-31 F,32-63 G,64-191 H  ~51 KB

    const int b    = blockIdx.y;
    const int n0   = blockIdx.x * 128;
    const int t    = threadIdx.x;
    const int wave = t >> 5;
    const int lane = t & 31;
    const int half = lane >> 4;
    const int l16  = lane & 15;
    const float* xb = x + (size_t)b * DIM * NTOK;

    // stage X^T: thread t owns n-row t; 8 coalesced b32 loads -> 1 b128 store
    #pragma unroll 2
    for (int cc = 0; cc < 16; ++cc) {
        pk8 d; pk2 e;
        #pragma unroll
        for (int j = 0; j < 4; ++j) {
            e.s.lo = (__bf16)xb[(size_t)(cc * 8 + 2 * j)     * NTOK + n0 + t];
            e.s.hi = (__bf16)xb[(size_t)(cc * 8 + 2 * j + 1) * NTOK + n0 + t];
            d.w[j] = e.u;
        }
        *(uint4*)&xsT[t][cc * 8] = d.q;
    }
    // stage W (bf16), folding log2e into WG
    #pragma unroll 2
    for (int i = 0; i < 24; ++i) {
        const int g  = t + 128 * i;
        const int r  = g >> 4;
        const int c0 = (g & 15) * 8;
        const float* wsrc; float sc = 1.0f;
        if (r < 32)       wsrc = WF + (size_t)r * DIM;
        else if (r < 64) { wsrc = WG + (size_t)(r - 32) * DIM; sc = LOG2E; }
        else              wsrc = WH + (size_t)(r - 64) * DIM;
        float4 w0 = *(const float4*)(wsrc + c0);
        float4 w1 = *(const float4*)(wsrc + c0 + 4);
        pk8 d; pk2 e;
        e.s.lo = (__bf16)(w0.x * sc); e.s.hi = (__bf16)(w0.y * sc); d.w[0] = e.u;
        e.s.lo = (__bf16)(w0.z * sc); e.s.hi = (__bf16)(w0.w * sc); d.w[1] = e.u;
        e.s.lo = (__bf16)(w1.x * sc); e.s.hi = (__bf16)(w1.y * sc); d.w[2] = e.u;
        e.s.lo = (__bf16)(w1.z * sc); e.s.hi = (__bf16)(w1.w * sc); d.w[3] = e.u;
        *(uint4*)&Wb[r][c0] = d.q;
    }
    __syncthreads();

    // compute: wave owns n-tiles {2w, 2w+1}; 12 o-tiles x 4 K-chunks
    v8f zero = {};
    const __bf16* xrow0 = &xsT[wave * 32 + l16][0];
    const __bf16* xrow1 = &xsT[wave * 32 + 16 + l16][0];

    for (int ot = 0; ot < 12; ++ot) {
        v8f acc0 = zero, acc1 = zero;
        const __bf16* wrow = &Wb[ot * 16 + l16][0];
        #pragma unroll
        for (int kc = 0; kc < 4; ++kc) {
            frag16 aw, bx;
            aw.u[0] = *(const uint4*)(wrow + kc * 32 + 16 * half);
            aw.u[1] = *(const uint4*)(wrow + kc * 32 + 16 * half + 8);
            bx.u[0] = *(const uint4*)(xrow0 + kc * 32 + 16 * half);
            bx.u[1] = *(const uint4*)(xrow0 + kc * 32 + 16 * half + 8);
            acc0 = __builtin_amdgcn_wmma_f32_16x16x32_bf16(
                false, aw.v, false, bx.v, (short)0, acc0, false, false);
            bx.u[0] = *(const uint4*)(xrow1 + kc * 32 + 16 * half);
            bx.u[1] = *(const uint4*)(xrow1 + kc * 32 + 16 * half + 8);
            acc1 = __builtin_amdgcn_wmma_f32_16x16x32_bf16(
                false, aw.v, false, bx.v, (short)0, acc1, false, false);
        }
        #pragma unroll
        for (int s = 0; s < 2; ++s) {
            const v8f& a = s ? acc1 : acc0;
            const int n = n0 + (wave * 2 + s) * 16 + l16;
            if (ot < 4) {             // f (ot 0,1) / g (ot 2,3): o-run of 8 -> b128
                pk8 d; pk2 e;
                e.s.lo = (__bf16)a[0]; e.s.hi = (__bf16)a[1]; d.w[0] = e.u;
                e.s.lo = (__bf16)a[2]; e.s.hi = (__bf16)a[3]; d.w[1] = e.u;
                e.s.lo = (__bf16)a[4]; e.s.hi = (__bf16)a[5]; d.w[2] = e.u;
                e.s.lo = (__bf16)a[6]; e.s.hi = (__bf16)a[7]; d.w[3] = e.u;
                __bf16* dst = (ot < 2) ? f_ws : g_ws;
                const int orow = (ot & 1) * 16 + 8 * half;
                *(uint4*)(dst + ((size_t)b * NTOK + n) * LAYER + orow) = d.q;
            } else {                  // h: transposed store, coalesced across lanes
                const int d0 = (ot - 4) * 16 + 8 * half;
                #pragma unroll
                for (int v = 0; v < 8; ++v)
                    h_wsT[((size_t)b * DIM + d0 + v) * NTOK + n] = (__bf16)a[v];
            }
        }
    }
}

// ---------------------------------------------------------------------------
// Phase 2: flash attention, transposed formulation, log2-domain softmax.
// ---------------------------------------------------------------------------
__global__ __launch_bounds__(128, 1) void attn_kernel(
    const float* __restrict__ x, const float* __restrict__ gamma,
    const __bf16* __restrict__ f_ws, const __bf16* __restrict__ g_ws,
    const __bf16* __restrict__ h_wsT, float* __restrict__ out)
{
    __shared__ __align__(16) __bf16 fch[2][32][FPAD];
    __shared__ __align__(16) __bf16 hT[2][DIM][FPAD];

    const int b    = blockIdx.y;
    const int t    = threadIdx.x;
    const int wave = t >> 5;
    const int lane = t & 31;
    const int half = lane >> 4;
    const int l16  = lane & 15;
    const int q0   = blockIdx.x * 64 + wave * 16;

    const __bf16* grow = g_ws + ((size_t)b * NTOK + q0 + l16) * LAYER;
    frag16 bg;
    bg.u[0] = *(const uint4*)(grow + 8 * half);
    bg.u[1] = *(const uint4*)(grow + 16 + 8 * half);

    v8f zero = {};
    v8f acc[8];
    #pragma unroll
    for (int nt = 0; nt < 8; ++nt) acc[nt] = zero;
    float mq = -__builtin_inff(), lq = 0.f;   // log2-domain stats

    const __bf16* fbase = f_ws + (size_t)b * NTOK * LAYER;
    const __bf16* hbase = h_wsT + ((size_t)b * DIM + t) * NTOK;
    const int fr_ = t >> 2, fq_ = (t & 3) * 8;

#ifdef HAVE_ASYNC_LDS
    async_cp_b128(fbase + (size_t)fr_ * LAYER + fq_, &fch[0][fr_][fq_]);
    #pragma unroll
    for (int i = 0; i < 4; ++i)
        async_cp_b128(hbase + i * 8, &hT[0][t][i * 8]);
    async_wait0();
#else
    *(uint4*)&fch[0][fr_][fq_] = *(const uint4*)(fbase + (size_t)fr_ * LAYER + fq_);
    {
        const uint4* hs = (const uint4*)hbase;
        uint4* hd = (uint4*)&hT[0][t][0];
        hd[0] = hs[0]; hd[1] = hs[1]; hd[2] = hs[2]; hd[3] = hs[3];
    }
#endif
    __syncthreads();

    for (int kc = 0; kc < NTOK; kc += 32) {
        const int  cur  = (kc >> 5) & 1;
        const bool have = (kc + 32) < NTOK;

#ifdef HAVE_ASYNC_LDS
        if (have) {
            async_cp_b128(fbase + (size_t)(kc + 32 + fr_) * LAYER + fq_,
                          &fch[cur ^ 1][fr_][fq_]);
            #pragma unroll
            for (int i = 0; i < 4; ++i)
                async_cp_b128(hbase + kc + 32 + i * 8, &hT[cur ^ 1][t][i * 8]);
        }
#else
        uint4 nf, nh[4];
        if (have) {
            nf = *(const uint4*)(fbase + (size_t)(kc + 32 + fr_) * LAYER + fq_);
            const uint4* hs = (const uint4*)(hbase + kc + 32);
            nh[0] = hs[0]; nh[1] = hs[1]; nh[2] = hs[2]; nh[3] = hs[3];
        }
#endif

        // ---- S^T = F_k · G_q^T (G pre-scaled by log2e) ----
        v8f c0, c1;
        {
            frag16 af;
            const __bf16* fr0 = &fch[cur][l16][0];
            af.u[0] = *(const uint4*)(fr0 + 8 * half);
            af.u[1] = *(const uint4*)(fr0 + 16 + 8 * half);
            c0 = __builtin_amdgcn_wmma_f32_16x16x32_bf16(
                false, af.v, false, bg.v, (short)0, zero, false, false);
            const __bf16* fr1 = &fch[cur][16 + l16][0];
            af.u[0] = *(const uint4*)(fr1 + 8 * half);
            af.u[1] = *(const uint4*)(fr1 + 16 + 8 * half);
            c1 = __builtin_amdgcn_wmma_f32_16x16x32_bf16(
                false, af.v, false, bg.v, (short)0, zero, false, false);
        }

        // ---- softmax (log2 domain): register tree + one cross-half exchange ----
        float a0 = fmaxf(c0[0], c0[1]), a1 = fmaxf(c0[2], c0[3]);
        float a2 = fmaxf(c0[4], c0[5]), a3 = fmaxf(c0[6], c0[7]);
        float b0 = fmaxf(c1[0], c1[1]), b1 = fmaxf(c1[2], c1[3]);
        float b2 = fmaxf(c1[4], c1[5]), b3 = fmaxf(c1[6], c1[7]);
        float mloc = fmaxf(fmaxf(fmaxf(a0, a1), fmaxf(a2, a3)),
                           fmaxf(fmaxf(b0, b1), fmaxf(b2, b3)));
        mloc = fmaxf(mloc, permx16_f32(mloc));
        float mnew = fmaxf(mq, mloc);
        float scl  = fast_exp2(mq - mnew);
        bool  resc = scl < 1.0f;

        float p0[8], p1[8];
        #pragma unroll
        for (int v = 0; v < 8; ++v) {
            p0[v] = fast_exp2(c0[v] - mnew);
            p1[v] = fast_exp2(c1[v] - mnew);
        }
        float s0 = (p0[0] + p0[1]) + (p0[2] + p0[3]);
        float s1 = (p0[4] + p0[5]) + (p0[6] + p0[7]);
        float s2 = (p1[0] + p1[1]) + (p1[2] + p1[3]);
        float s3 = (p1[4] + p1[5]) + (p1[6] + p1[7]);
        lq = lq * scl + ((s0 + s1) + (s2 + s3));
        mq = mnew;

        // ---- B-fragment of P^T in registers ----
        frag16 bp;
        #pragma unroll
        for (int w = 0; w < 4; ++w) {
            pk2 e0; e0.s.lo = (__bf16)p0[2 * w]; e0.s.hi = (__bf16)p0[2 * w + 1];
            pk2 e1; e1.s.lo = (__bf16)p1[2 * w]; e1.s.hi = (__bf16)p1[2 * w + 1];
            unsigned x0 = permx16_u32(e0.u);
            unsigned x1 = permx16_u32(e1.u);
            bp.w32[w]     = half ? x1   : e0.u;
            bp.w32[w + 4] = half ? e1.u : x0;
        }

        if (__any((int)resc)) {
            #pragma unroll
            for (int nt = 0; nt < 8; ++nt)
                #pragma unroll
                for (int v = 0; v < 8; ++v)
                    acc[nt][v] *= scl;
        }

        // ---- O^T += H^T · P^T ----
        #pragma unroll
        for (int nt = 0; nt < 8; ++nt) {
            frag16 ah;
            const __bf16* hr = &hT[cur][nt * 16 + l16][0];
            ah.u[0] = *(const uint4*)(hr + 16 * half);
            ah.u[1] = *(const uint4*)(hr + 16 * half + 8);
            acc[nt] = __builtin_amdgcn_wmma_f32_16x16x32_bf16(
                false, ah.v, false, bp.v, (short)0, acc[nt], false, false);
        }

#ifdef HAVE_ASYNC_LDS
        if (have) async_wait0();
#else
        if (have) {
            *(uint4*)&fch[cur ^ 1][fr_][fq_] = nf;
            uint4* hd = (uint4*)&hT[cur ^ 1][t][0];
            hd[0] = nh[0]; hd[1] = nh[1]; hd[2] = nh[2]; hd[3] = nh[3];
        }
#endif
        __syncthreads();
    }

    // ---- epilogue ----
    const float gm  = gamma[0];
    const float inv = 1.0f / (lq + permx16_f32(lq));
    const size_t rowbase = (size_t)b * DIM * NTOK + (size_t)(q0 + l16) * DIM;
    #pragma unroll
    for (int nt = 0; nt < 8; ++nt) {
        const size_t idx = rowbase + nt * 16 + 8 * half;
        float4 xi0 = *(const float4*)(x + idx);
        float4 xi1 = *(const float4*)(x + idx + 4);
        float4 o0, o1;
        o0.x = gm * (acc[nt][0] * inv) + xi0.x;
        o0.y = gm * (acc[nt][1] * inv) + xi0.y;
        o0.z = gm * (acc[nt][2] * inv) + xi0.z;
        o0.w = gm * (acc[nt][3] * inv) + xi0.w;
        o1.x = gm * (acc[nt][4] * inv) + xi1.x;
        o1.y = gm * (acc[nt][5] * inv) + xi1.y;
        o1.z = gm * (acc[nt][6] * inv) + xi1.z;
        o1.w = gm * (acc[nt][7] * inv) + xi1.w;
        *(float4*)(out + idx)     = o0;
        *(float4*)(out + idx + 4) = o1;
    }
}

// ---------------------------------------------------------------------------
extern "C" void kernel_launch(void* const* d_in, const int* in_sizes, int n_in,
                              void* d_out, int out_size, void* d_ws, size_t ws_size,
                              hipStream_t stream) {
    (void)in_sizes; (void)n_in; (void)out_size; (void)ws_size;
    const float* x     = (const float*)d_in[0];
    const float* WF    = (const float*)d_in[1];
    const float* WG    = (const float*)d_in[2];
    const float* WH    = (const float*)d_in[3];
    const float* gamma = (const float*)d_in[4];

    __bf16* f_ws  = (__bf16*)d_ws;                              // 2 MB
    __bf16* g_ws  = f_ws + (size_t)BATCH * NTOK * LAYER;        // 2 MB
    __bf16* h_wsT = g_ws + (size_t)BATCH * NTOK * LAYER;        // 8 MB
    float* out = (float*)d_out;

    dim3 pgrid(NTOK / 128, BATCH);
    proj_kernel<<<pgrid, 128, 0, stream>>>(x, WF, WG, WH, f_ws, g_ws, h_wsT);

    dim3 agrid(NTOK / 64, BATCH);
    attn_kernel<<<agrid, 128, 0, stream>>>(x, gamma, f_ws, g_ws, h_wsT, out);
}
